// KANLayer_962072674577
// MI455X (gfx1250) — compile-verified, matrix operand order checked by reading, not run on hardware
//
#include <hip/hip_runtime.h>

typedef __attribute__((ext_vector_type(16))) _Float16 v16h;
typedef __attribute__((ext_vector_type(8)))  float    v8f;

#define B_DIM 8192
#define I_DIM 512
#define O_DIM 512
#define KSTEPS (I_DIM / 32)          // 16 K-blocks of 32
#define NTILES (O_DIM / 16)          // 32 column tiles
#define NT_PER_WAVE 4                // each wave: 16 x 64 output strip
#define NSUPER (O_DIM / (16 * NT_PER_WAVE))  // 8

// ---------------------------------------------------------------------------
// Prep kernel 1: pack W_eff[k][i][o] = edge_w[i][o][k] * comb_w[i][o] as f16
// in the wave32 WMMA B-fragment layout:
//   fragment id = (k*KSTEPS + iblk)*NTILES + ntile, 32 lanes x 16 f16 each.
//   lane L: column n = ntile*16 + (L&15); k-slots (per dword j=0..7):
//   kk = (j<4?0:16) + (L>=16?8:0) + 2*(j%4), pair (kk, kk+1).
// One thread produces one packed dword (two f16).
// ---------------------------------------------------------------------------
__global__ void kan_pack_w(const float* __restrict__ edge_w,
                           const float* __restrict__ comb_w,
                           unsigned int* __restrict__ wp) {
  int tid   = blockIdx.x * blockDim.x + threadIdx.x;   // dword id
  int j     = tid & 7;
  int lane  = (tid >> 3) & 31;
  int frag  = tid >> 8;
  int ntile = frag & (NTILES - 1);
  int iblk  = (frag >> 5) & (KSTEPS - 1);
  int k     = frag >> 9;                               // 0..2
  int n     = ntile * 16 + (lane & 15);
  int hi    = (lane >> 4) & 1;
  int kk    = ((j & 4) << 2) + (hi << 3) + ((j & 3) << 1);
  int i     = iblk * 32 + kk;

  float v0 = edge_w[(size_t)(i * O_DIM + n) * 3 + k]       * comb_w[i * O_DIM + n];
  float v1 = edge_w[(size_t)((i + 1) * O_DIM + n) * 3 + k] * comb_w[(i + 1) * O_DIM + n];

  union { _Float16 h[2]; unsigned int u; } p;
  p.h[0] = (_Float16)v0;
  p.h[1] = (_Float16)v1;
  wp[tid] = p.u;
}

// ---------------------------------------------------------------------------
// Prep kernel 2: bias[o] = sum_i comb_w[i][o] * edge_b[i][o]   (f32)
// ---------------------------------------------------------------------------
__global__ void kan_bias(const float* __restrict__ comb_w,
                         const float* __restrict__ edge_b,
                         float* __restrict__ bias) {
  int o = blockIdx.x * blockDim.x + threadIdx.x;
  if (o >= O_DIM) return;
  float s = 0.0f;
  for (int i = 0; i < I_DIM; ++i)
    s += comb_w[i * O_DIM + o] * edge_b[i * O_DIM + o];
  bias[o] = s;
}

// ---------------------------------------------------------------------------
// Main kernel: fused cubic-basis GEMM via v_wmma_f32_16x16x32_f16.
// One wave computes a 16x64 output strip (4 accumulators).
// A-fragments (x, x^2, x^3) built on the fly from float2 loads of x.
// ---------------------------------------------------------------------------
__global__ void __launch_bounds__(256)
kan_wmma(const float* __restrict__ x,
         const v16h*  __restrict__ Bp,
         const float* __restrict__ bias,
         float* __restrict__ out) {
  const int lane   = threadIdx.x & 31;
  const int wid    = blockIdx.x * (blockDim.x >> 5) + (threadIdx.x >> 5);
  const int nsuper = wid & (NSUPER - 1);
  const int mtile  = wid >> 3;                 // log2(NSUPER)
  const int hi     = lane >> 4;                // 0 / 1 lane half
  const int mrow   = mtile * 16 + (lane & 15);
  const float* __restrict__ xrow = x + (size_t)mrow * I_DIM;

  v8f acc[NT_PER_WAVE] = {};

  for (int iblk = 0; iblk < KSTEPS; ++iblk) {
    // Build the three basis A-fragments (16x32 f16, wave32 layout).
    v16h a1, a2, a3;
#pragma unroll
    for (int j = 0; j < 8; ++j) {
      const int kk = ((j & 4) << 2) + (hi << 3) + ((j & 3) << 1);
      const float2 t = *(const float2*)(xrow + iblk * 32 + kk);
      const float s0 = t.x,      s1 = t.y;
      const float q0 = s0 * s0,  q1 = s1 * s1;
      const float c0 = q0 * s0,  c1 = q1 * s1;
      a1[2 * j] = (_Float16)s0;  a1[2 * j + 1] = (_Float16)s1;
      a2[2 * j] = (_Float16)q0;  a2[2 * j + 1] = (_Float16)q1;
      a3[2 * j] = (_Float16)c0;  a3[2 * j + 1] = (_Float16)c1;
    }

    const size_t plane = (size_t)KSTEPS * NTILES * 32;   // per-k fragment plane
    const v16h* __restrict__ base =
        Bp + ((size_t)iblk * NTILES + nsuper * NT_PER_WAVE) * 32 + lane;

#pragma unroll
    for (int t = 0; t < NT_PER_WAVE; ++t) {
      v16h b1 = base[t * 32];
      v16h b2 = base[plane + t * 32];
      v16h b3 = base[2 * plane + t * 32];
      acc[t] = __builtin_amdgcn_wmma_f32_16x16x32_f16(
          false, a1, false, b1, (short)0, acc[t], false, false);
      acc[t] = __builtin_amdgcn_wmma_f32_16x16x32_f16(
          false, a2, false, b2, (short)0, acc[t], false, false);
      acc[t] = __builtin_amdgcn_wmma_f32_16x16x32_f16(
          false, a3, false, b3, (short)0, acc[t], false, false);
    }
  }

  // Write-out: C/D layout — VGPR r: lanes 0-15 -> (M=r, N=lane),
  // lanes 16-31 -> (M=r+8, N=lane-16).
#pragma unroll
  for (int t = 0; t < NT_PER_WAVE; ++t) {
    const int n  = (nsuper * NT_PER_WAVE + t) * 16 + (lane & 15);
    const float bv = bias[n];
#pragma unroll
    for (int r = 0; r < 8; ++r) {
      const int m = mtile * 16 + hi * 8 + r;
      out[(size_t)m * O_DIM + n] = acc[t][r] + bv;
    }
  }
}

// ---------------------------------------------------------------------------
// Launch: prep (pack W + bias) then fused WMMA GEMM, all stream-ordered.
// Workspace: [0, 1.5MB) packed f16 W, then 512 f32 bias.
// ---------------------------------------------------------------------------
extern "C" void kernel_launch(void* const* d_in, const int* in_sizes, int n_in,
                              void* d_out, int out_size, void* d_ws, size_t ws_size,
                              hipStream_t stream) {
  const float* x      = (const float*)d_in[0];
  const float* edge_w = (const float*)d_in[1];
  const float* edge_b = (const float*)d_in[2];
  const float* comb_w = (const float*)d_in[3];
  float* out = (float*)d_out;

  unsigned char* ws = (unsigned char*)d_ws;
  unsigned int* wp  = (unsigned int*)ws;
  float* bias = (float*)(ws + (size_t)3 * I_DIM * O_DIM * sizeof(_Float16));

  // 3*16*32 fragments * 32 lanes * 8 dwords = 393216 dwords
  kan_pack_w<<<1536, 256, 0, stream>>>(edge_w, comb_w, wp);
  kan_bias<<<2, 256, 0, stream>>>(comb_w, edge_b, bias);

  // 512 m-tiles * 8 n-supertiles = 4096 waves -> 512 blocks of 8 waves
  kan_wmma<<<512, 256, 0, stream>>>(x, (const v16h*)wp, bias, out);
}
